// leap_op_rank_17592186044589
// MI455X (gfx1250) — matrix operand order repaired and checked by reading, not compiled, not run on hardware
//
#include <hip/hip_runtime.h>
#include <hip/hip_bf16.h>

typedef __attribute__((ext_vector_type(16))) __bf16 v16bf;
typedef __attribute__((ext_vector_type(8)))  float  v8f;

#define HID 256
constexpr int cNE   = 20000;
constexpr int cNR   = 250;
constexpr int cR2   = 500;
constexpr int cT    = 3;
constexpr int cE    = 150000;
constexpr int cM    = 200000;
constexpr int cNNZE = 320000;
constexpr int cNNZR = 8000;
constexpr int cCH   = 50;
constexpr int cNT   = 2000;
constexpr int cNTT  = 4000;
constexpr int cKFC  = cCH * HID;   // 12800

// ---- WMMA helpers -------------------------------------------------------
// A (16x32 bf16): lane L holds row M=L%16; element e -> K per ISA table.
__device__ __forceinline__ int akof(int lane, int e) {
  int vg   = e >> 1;
  int base = (vg >= 4) ? 16 : 0;
  return base + ((lane & 16) ? 8 : 0) + (vg & 3) * 2 + (e & 1);
}

__device__ __forceinline__ v8f zero8() {
  v8f z;
#pragma unroll
  for (int i = 0; i < 8; i++) z[i] = 0.0f;
  return z;
}

__device__ __forceinline__ v8f wmma_bf16(v16bf a, v16bf b, v8f c) {
  return __builtin_amdgcn_wmma_f32_16x16x32_bf16(false, a, false, b,
                                                 (short)0, c, false, false);
}

// Load A fragment from LDS; element pairs (2v,2v+1) are K-adjacent -> b32 reads.
__device__ __forceinline__ v16bf load_a_lds(const __bf16* base, int stride,
                                            int lane, int k0) {
  union { v16bf v; unsigned u[8]; } r;
  int m = lane & 15;
  const unsigned* p32 = (const unsigned*)(base + m * stride + k0);
#pragma unroll
  for (int vg = 0; vg < 8; vg++) r.u[vg] = p32[akof(lane, 2 * vg) >> 1];
  return r.v;
}

__device__ __forceinline__ v16bf load_b_g(const __bf16* p, int stride,
                                          int lane, int k0) {
  v16bf r;
#pragma unroll
  for (int e = 0; e < 16; e++)
    r[e] = p[(size_t)(k0 + akof(lane, e)) * stride + (lane & 15)];
  return r;
}

// ---- elementwise / utility kernels --------------------------------------
__global__ void k_zero(float* p, int n) {
  int i = blockIdx.x * 256 + threadIdx.x;
  if (i < n) p[i] = 0.0f;
}

__global__ void k_f2bf(const float* __restrict__ s, __bf16* __restrict__ d, int n) {
  int i = blockIdx.x * 256 + threadIdx.x;
  if (i < n) d[i] = (__bf16)s[i];
}

// W (N x K row-major, f32) -> WT (K x N, bf16)
__global__ void k_wt2bf(const float* __restrict__ W, __bf16* __restrict__ WT,
                        int N, int K) {
  int idx = blockIdx.x * 256 + threadIdx.x;
  if (idx >= N * K) return;
  int n = idx / K, k = idx % K;
  WT[k * N + n] = (__bf16)W[idx];
}

// seg-mean scatter for x_mean: xsum[r_ids[j]] += ent[r_to_e[j]]
__global__ void k_scatter_xmean(const float* __restrict__ ent,
                                const int* __restrict__ rtoe,
                                const int* __restrict__ rids,
                                float* xsum, float* xcnt) {
  long long idx = (long long)blockIdx.x * 256 + threadIdx.x;
  if (idx >= (long long)cM * HID) return;
  int j = (int)(idx >> 8);
  int h = (int)(idx & 255);
  int row = rids[j];
  int e   = rtoe[j];
  atomicAdd(&xsum[row * HID + h], ent[e * HID + h]);
  if (h == 0) atomicAdd(&xcnt[row], 1.0f);
}

__global__ void k_build_xin(const float* __restrict__ rel,
                            const float* __restrict__ xsum,
                            const float* __restrict__ xcnt,
                            float* __restrict__ xin) {
  int idx = blockIdx.x * 256 + threadIdx.x;
  if (idx >= cR2 * 2 * HID) return;
  int r = idx >> 9;
  int c = idx & 511;
  float v;
  if (c < HID) v = rel[r * HID + c];
  else         v = xsum[r * HID + (c - HID)] / fmaxf(xcnt[r], 1.0f);
  xin[idx] = v;
}

// ---- GEMM: C[M x N] = A[M x K] @ WT + bias; WT is (K x N) bf16 ----------
// LDS-staged A (rows >= M zero-padded), vector B loads, dual accumulators.
__global__ void k_gemm_bias(const float* __restrict__ A, int Mr, int K,
                            const __bf16* __restrict__ WT,
                            const float* __restrict__ bias,
                            float* __restrict__ C, int N) {
  extern __shared__ __align__(16) __bf16 As[];  // 16 * K
  int rb  = blockIdx.x * 16;
  int tid = threadIdx.x;
  for (int i = tid; i < 16 * K; i += 256) {
    int r = i / K, k = i % K;
    int gr = rb + r;
    As[i] = (gr < Mr) ? (__bf16)A[gr * K + k] : (__bf16)0.0f;
  }
  __syncthreads();
  int wave = tid >> 5, lane = tid & 31;
  int ntn  = N >> 4;
  const __bf16* bp0 = WT;
  for (int nt = wave; nt < ntn; nt += 8) {
    int n0  = nt * 16;
    const __bf16* bp = bp0 + n0;
    v8f acc0 = zero8(), acc1 = zero8();
    for (int k0 = 0; k0 < K; k0 += 64) {
      v16bf a0 = load_a_lds(As, K, lane, k0);
      v16bf b0 = load_b_g(bp, N, lane, k0);
      acc0 = wmma_bf16(a0, b0, acc0);
      v16bf a1 = load_a_lds(As, K, lane, k0 + 32);
      v16bf b1 = load_b_g(bp, N, lane, k0 + 32);
      acc1 = wmma_bf16(a1, b1, acc1);
    }
    int ncol = n0 + (lane & 15);
    int mh   = (lane & 16) ? 8 : 0;
    float bv = bias[ncol];
#pragma unroll
    for (int r = 0; r < 8; r++) {
      int gr = rb + r + mh;
      if (gr < Mr) C[gr * N + ncol] = acc0[r] + acc1[r] + bv;
    }
  }
}

__global__ void k_gru_gates(const float* __restrict__ gi,
                            const float* __restrict__ gh,
                            const float* __restrict__ relin,
                            float* __restrict__ relout) {
  int idx = blockIdx.x * 256 + threadIdx.x;
  if (idx >= cR2 * HID) return;
  int r = idx >> 8, h = idx & 255;
  const float* gir = gi + r * 3 * HID;
  const float* ghr = gh + r * 3 * HID;
  float ir = gir[h], iz = gir[HID + h], in = gir[2 * HID + h];
  float hr = ghr[h], hz = ghr[HID + h], hn = ghr[2 * HID + h];
  float rr = 1.0f / (1.0f + __expf(-(ir + hr)));
  float zz = 1.0f / (1.0f + __expf(-(iz + hz)));
  float nn = tanhf(in + rr * hn);
  relout[idx] = (1.0f - zz) * nn + zz * relin[idx];
}

// ---- msg GEMM with gathered A rows, scatter-add epilogue ----------------
// Each wave computes its two n-tiles simultaneously: A-frag reuse + 2 chains.
__global__ void k_msg(const float* __restrict__ ent,
                      const float* __restrict__ rel,
                      const int* __restrict__ esrc,
                      const int* __restrict__ erel,
                      const int* __restrict__ edst,
                      const __bf16* __restrict__ aggWbf,
                      float* __restrict__ csum) {
  __shared__ __align__(16) __bf16 As[16][HID];
  __shared__ int dstS[16];
  int eb  = blockIdx.x * 16;
  int tid = threadIdx.x;
  int row = tid >> 4;
  int cb  = (tid & 15) * 16;
  int s  = esrc[eb + row];
  int rl = erel[eb + row];
  const float* ep = ent + (long long)s * HID;
  const float* rp = rel + (long long)rl * HID;
#pragma unroll
  for (int c = 0; c < 16; c++) As[row][cb + c] = (__bf16)(ep[cb + c] + rp[cb + c]);
  if (tid < 16) dstS[tid] = edst[eb + tid];
  __syncthreads();
  int wave = tid >> 5, lane = tid & 31;
  int n0 = wave * 16, n1 = (wave + 8) * 16;
  const __bf16* bp0 = aggWbf + n0;
  const __bf16* bp1 = aggWbf + n1;
  v8f acc0 = zero8(), acc1 = zero8();
#pragma unroll
  for (int k0 = 0; k0 < HID; k0 += 32) {
    v16bf a  = load_a_lds(&As[0][0], HID, lane, k0);
    v16bf b0 = load_b_g(bp0, HID, lane, k0);
    v16bf b1 = load_b_g(bp1, HID, lane, k0);
    acc0 = wmma_bf16(a, b0, acc0);
    acc1 = wmma_bf16(a, b1, acc1);
  }
  int mh = (lane & 16) ? 8 : 0;
  int nc0 = n0 + (lane & 15), nc1 = n1 + (lane & 15);
#pragma unroll
  for (int r = 0; r < 8; r++) {
    int d = dstS[r + mh] * HID;
    atomicAdd(&csum[d + nc0], acc0[r]);
    atomicAdd(&csum[d + nc1], acc1[r]);
  }
}

__global__ void k_count(const int* __restrict__ edst, float* ccnt) {
  int j = blockIdx.x * 256 + threadIdx.x;
  if (j < cE) atomicAdd(&ccnt[edst[j]], 1.0f);
}

// ---- gate GEMM: tw = sigmoid(ent@tg_W+b); ent' = tw*relu(mean)+(1-tw)*ent
__global__ void k_gate(const float* __restrict__ ent_in,
                       const __bf16* __restrict__ tgWbf,
                       const float* __restrict__ tgb,
                       const float* __restrict__ csum,
                       const float* __restrict__ ccnt,
                       float* __restrict__ ent_out) {
  __shared__ __align__(16) __bf16 As[16][HID];
  int rb  = blockIdx.x * 16;
  int tid = threadIdx.x;
  int row = tid >> 4;
  int cb  = (tid & 15) * 16;
  const float* ep = ent_in + (long long)(rb + row) * HID;
#pragma unroll
  for (int c = 0; c < 16; c++) As[row][cb + c] = (__bf16)ep[cb + c];
  __syncthreads();
  int wave = tid >> 5, lane = tid & 31;
  int n0 = wave * 16, n1 = (wave + 8) * 16;
  const __bf16* bp0 = tgWbf + n0;
  const __bf16* bp1 = tgWbf + n1;
  v8f acc0 = zero8(), acc1 = zero8();
#pragma unroll
  for (int k0 = 0; k0 < HID; k0 += 32) {
    v16bf a  = load_a_lds(&As[0][0], HID, lane, k0);
    v16bf b0 = load_b_g(bp0, HID, lane, k0);
    v16bf b1 = load_b_g(bp1, HID, lane, k0);
    acc0 = wmma_bf16(a, b0, acc0);
    acc1 = wmma_bf16(a, b1, acc1);
  }
  int mh = (lane & 16) ? 8 : 0;
#pragma unroll
  for (int t = 0; t < 2; t++) {
    int ncol = (t ? n1 : n0) + (lane & 15);
    v8f acc  = t ? acc1 : acc0;
    float bv = tgb[ncol];
#pragma unroll
    for (int r = 0; r < 8; r++) {
      int gr   = rb + r + mh;
      float tw = 1.0f / (1.0f + __expf(-(acc[r] + bv)));
      float cu = fmaxf(csum[gr * HID + ncol] / fmaxf(ccnt[gr], 1.0f), 0.0f);
      ent_out[gr * HID + ncol] = tw * cu + (1.0f - tw) * ent_in[gr * HID + ncol];
    }
  }
}

// ---- hypergraph propagation ---------------------------------------------
__global__ void k_hscatter(int nnz, const float* __restrict__ val,
                           const int* __restrict__ rows,
                           const int* __restrict__ cols,
                           const float* __restrict__ x, float* __restrict__ y) {
  long long idx = (long long)blockIdx.x * 256 + threadIdx.x;
  if (idx >= (long long)nnz * HID) return;
  int j = (int)(idx >> 8);
  int h = (int)(idx & 255);
  atomicAdd(&y[rows[j] * HID + h], val[j] * x[cols[j] * HID + h]);
}

__global__ void k_l2res(const float* __restrict__ y,
                        const float* __restrict__ xin,
                        float* __restrict__ xout) {
  __shared__ float red[256];
  int r   = blockIdx.x;
  int tid = threadIdx.x;
  float v = y[r * HID + tid];
  red[tid] = v * v;
  __syncthreads();
  for (int s2 = 128; s2 > 0; s2 >>= 1) {
    if (tid < s2) red[tid] += red[tid + s2];
    __syncthreads();
  }
  float nrm = fmaxf(sqrtf(red[0]), 1e-12f);
  xout[r * HID + tid] = v / nrm + xin[r * HID + tid];
}

// ---- decoder -------------------------------------------------------------
__global__ void k_feat(const int* __restrict__ trip,
                       const float* __restrict__ ent,
                       const float* __restrict__ rel,
                       const float* __restrict__ sent,
                       float* __restrict__ feat) {
  int idx = blockIdx.x * 256 + threadIdx.x;
  if (idx >= cNTT * HID) return;
  int n = idx >> 8, h = idx & 255;
  int e, rr, srow;
  if (n < cNT) { e = trip[n * 3 + 0]; rr = trip[n * 3 + 1]; srow = n; }
  else { int m2 = n - cNT; e = trip[m2 * 3 + 2]; rr = trip[m2 * 3 + 1] + cNR; srow = m2; }
  float* fr = feat + n * 3 * HID;
  fr[h]            = ent[e * HID + h];
  fr[HID + h]      = rel[rr * HID + h];
  fr[2 * HID + h]  = sent[srow * HID + h];
}

// fused: conv1d(3->50, k=3, pad 1) + relu + reshape + fc(12800->256) + relu
__global__ void k_convfc(const float* __restrict__ feat,
                         const float* __restrict__ convw,
                         const float* __restrict__ convb,
                         const __bf16* __restrict__ fcWbf,
                         const float* __restrict__ fcb,
                         float* __restrict__ q, __bf16* __restrict__ qbf) {
  __shared__ __align__(16) __bf16 F[16][3 * HID];     // 24 KB
  __shared__ __align__(16) __bf16 Ach[16][512];       // 16 KB
  __shared__ float CW[cCH * 9];
  __shared__ float CB[cCH];
  int nb  = blockIdx.x * 16;
  int tid = threadIdx.x;
  for (int i = tid; i < 16 * 3 * HID; i += 256)
    F[i / (3 * HID)][i % (3 * HID)] =
        (__bf16)feat[(nb + i / (3 * HID)) * 3 * HID + (i % (3 * HID))];
  for (int i = tid; i < cCH * 9; i += 256) CW[i] = convw[i];
  if (tid < cCH) CB[tid] = convb[tid];
  __syncthreads();
  int wave = tid >> 5, lane = tid & 31;
  v8f acc0 = zero8(), acc1 = zero8();
  const __bf16* bpl = fcWbf + wave * 16;
  for (int kc = 0; kc < cKFC; kc += 512) {
    // build conv-activation chunk on the fly; paired h -> packed b32 stores
    for (int i = tid; i < 16 * 256; i += 256) {
      int m  = i >> 8;
      int p  = i & 255;
      int kk = kc + p * 2;
      int c  = kk >> 8;
      int h  = kk & 255;              // even, h+1 <= 255
      float s0 = CB[c], s1 = CB[c];
#pragma unroll
      for (int ic = 0; ic < 3; ic++) {
        const float* w3 = &CW[c * 9 + ic * 3];
        float fm1 = (h > 0)       ? (float)F[m][ic * HID + h - 1] : 0.0f;
        float f0  = (float)F[m][ic * HID + h];
        float f1  = (float)F[m][ic * HID + h + 1];
        float f2  = (h + 2 < 256) ? (float)F[m][ic * HID + h + 2] : 0.0f;
        s0 += fm1 * w3[0] + f0 * w3[1] + f1 * w3[2];
        s1 += f0  * w3[0] + f1 * w3[1] + f2 * w3[2];
      }
      union { __bf16 b[2]; unsigned u; } pk;
      pk.b[0] = (__bf16)fmaxf(s0, 0.0f);
      pk.b[1] = (__bf16)fmaxf(s1, 0.0f);
      *(unsigned*)&Ach[m][p * 2] = pk.u;
    }
    __syncthreads();
    const __bf16* bpc = bpl + (size_t)kc * HID;
#pragma unroll
    for (int k0 = 0; k0 < 512; k0 += 32) {
      v16bf a  = load_a_lds(&Ach[0][0], 512, lane, k0);
      v16bf b0 = load_b_g(bpc, HID, lane, k0);
      v16bf b1 = load_b_g(bpc + 128, HID, lane, k0);
      acc0 = wmma_bf16(a, b0, acc0);
      acc1 = wmma_bf16(a, b1, acc1);
    }
    __syncthreads();
  }
  int mh = (lane & 16) ? 8 : 0;
#pragma unroll
  for (int t = 0; t < 2; t++) {
    int n0   = (wave + 8 * t) * 16;
    int ncol = n0 + (lane & 15);
    v8f acc  = t ? acc1 : acc0;
    float bv = fcb[ncol];
#pragma unroll
    for (int r = 0; r < 8; r++) {
      int gr  = nb + r + mh;
      float v = fmaxf(acc[r] + bv, 0.0f);
      q[gr * HID + ncol]   = v;
      qbf[gr * HID + ncol] = (__bf16)v;
    }
  }
}

__global__ void k_entT(const float* __restrict__ ent, __bf16* __restrict__ entT) {
  int idx = blockIdx.x * 256 + threadIdx.x;
  if (idx >= cNE * HID) return;
  int j = idx >> 8, h = idx & 255;
  entT[h * cNE + j] = (__bf16)ent[j * HID + h];
}

// scores = q @ ent^T fused with online logsumexp over 20000 columns
__global__ void k_scores(const __bf16* __restrict__ qbf,
                         const __bf16* __restrict__ entT,
                         float* __restrict__ logz) {
  __shared__ __align__(16) __bf16 Qs[16][HID];   // 8 KB, linear copy target
  __shared__ float red[8][16][2];
  int nb  = blockIdx.x * 16;
  int tid = threadIdx.x;
  // async global->LDS copy of the 16 q rows (bf16, no conversion needed)
  {
    unsigned ldsBase = (unsigned)(unsigned long long)(uintptr_t)&Qs[0][0];
    unsigned long long gbase =
        (unsigned long long)(uintptr_t)(qbf + (size_t)nb * HID);
#pragma unroll
    for (int i = 0; i < 2; i++) {
      unsigned chunk = tid + i * 256;              // 512 x 16B = 8 KB
      unsigned loff  = ldsBase + chunk * 16;
      unsigned long long ga = gbase + (unsigned long long)chunk * 16;
      asm volatile("global_load_async_to_lds_b128 %0, %1, off"
                   :: "v"(loff), "v"(ga) : "memory");
    }
    asm volatile("s_wait_asynccnt 0" ::: "memory");
  }
  __syncthreads();
  int wave = tid >> 5, lane = tid & 31;
  float Mx[8], Sx[8];
#pragma unroll
  for (int r = 0; r < 8; r++) { Mx[r] = -1e30f; Sx[r] = 0.0f; }
  const __bf16* bpl = entT;
  for (int jt = wave; jt < cNE / 16; jt += 8) {
    int j0  = jt * 16;
    // prefetch next tile's B panel (one line per lane)
    if (jt + 8 < cNE / 16)
      __builtin_prefetch(entT + (size_t)lane * cNE + (jt + 8) * 16, 0, 0);
    const __bf16* bp = bpl + j0;
    v8f acc0 = zero8(), acc1 = zero8();
#pragma unroll
    for (int k0 = 0; k0 < HID; k0 += 64) {
      v16bf a0 = load_a_lds(&Qs[0][0], HID, lane, k0);
      v16bf b0 = load_b_g(bp, cNE, lane, k0);
      acc0 = wmma_bf16(a0, b0, acc0);
      v16bf a1 = load_a_lds(&Qs[0][0], HID, lane, k0 + 32);
      v16bf b1 = load_b_g(bp, cNE, lane, k0 + 32);
      acc1 = wmma_bf16(a1, b1, acc1);
    }
#pragma unroll
    for (int r = 0; r < 8; r++) {
      float v  = acc0[r] + acc1[r];
      float tm = v;
      for (int off = 1; off < 16; off <<= 1)
        tm = fmaxf(tm, __shfl_xor(tm, off, 32));
      float e = __expf(v - tm);
      for (int off = 1; off < 16; off <<= 1) e += __shfl_xor(e, off, 32);
      float nm = fmaxf(Mx[r], tm);
      Sx[r] = Sx[r] * __expf(Mx[r] - nm) + e * __expf(tm - nm);
      Mx[r] = nm;
    }
  }
  if ((lane & 15) == 0) {
    int hi = (lane >> 4) * 8;
#pragma unroll
    for (int r = 0; r < 8; r++) {
      red[wave][hi + r][0] = Mx[r];
      red[wave][hi + r][1] = Sx[r];
    }
  }
  __syncthreads();
  if (tid < 16) {
    float gm = -1e30f, gs = 0.0f;
    for (int w = 0; w < 8; w++) {
      float m_ = red[w][tid][0], s_ = red[w][tid][1];
      float nm = fmaxf(gm, m_);
      gs = gs * __expf(gm - nm) + s_ * __expf(m_ - nm);
      gm = nm;
    }
    logz[nb + tid] = gm + __logf(gs);
  }
}

__global__ void k_tscore(const float* __restrict__ q,
                         const float* __restrict__ ent,
                         const int* __restrict__ trip,
                         float* __restrict__ tsc) {
  __shared__ float red[256];
  int n   = blockIdx.x;
  int tid = threadIdx.x;
  int tgt = (n < cNT) ? trip[n * 3 + 2] : trip[(n - cNT) * 3 + 0];
  red[tid] = q[n * HID + tid] * ent[tgt * HID + tid];
  __syncthreads();
  for (int s2 = 128; s2 > 0; s2 >>= 1) {
    if (tid < s2) red[tid] += red[tid + s2];
    __syncthreads();
  }
  if (tid == 0) tsc[n] = red[0];
}

__global__ void k_final(const float* __restrict__ logz,
                        const float* __restrict__ tsc, float* out) {
  __shared__ float red[256];
  int tid = threadIdx.x;
  float s = 0.0f;
  for (int i = tid; i < cNTT; i += 256) s += logz[i] - tsc[i];
  red[tid] = s;
  __syncthreads();
  for (int s2 = 128; s2 > 0; s2 >>= 1) {
    if (tid < s2) red[tid] += red[tid + s2];
    __syncthreads();
  }
  if (tid == 0) out[0] = red[0] / (float)cNTT;
}

// ---- host side -----------------------------------------------------------
static inline int cdiv(long long a, long long b) { return (int)((a + b - 1) / b); }

extern "C" void kernel_launch(void* const* d_in, const int* in_sizes, int n_in,
                              void* d_out, int out_size, void* d_ws, size_t ws_size,
                              hipStream_t stream) {
  (void)in_sizes; (void)n_in; (void)out_size; (void)ws_size;
  const float* dyn   = (const float*)d_in[0];
  const float* erel0 = (const float*)d_in[1];
  const float* Wih   = (const float*)d_in[2];
  const float* Whh   = (const float*)d_in[3];
  const float* bih   = (const float*)d_in[4];
  const float* bhh   = (const float*)d_in[5];
  const float* aggW  = (const float*)d_in[6];
  const float* tgW   = (const float*)d_in[7];
  const float* tgb   = (const float*)d_in[8];
  const float* hev   = (const float*)d_in[9];
  const float* hrv   = (const float*)d_in[10];
  const float* convw = (const float*)d_in[11];
  const float* convb = (const float*)d_in[12];
  const float* fcW   = (const float*)d_in[13];
  const float* fcb   = (const float*)d_in[14];
  const float* sent  = (const float*)d_in[15];
  const int* esrc = (const int*)d_in[16];
  const int* erel = (const int*)d_in[17];
  const int* edst = (const int*)d_in[18];
  const int* rtoe = (const int*)d_in[19];
  const int* rids = (const int*)d_in[20];
  const int* herw = (const int*)d_in[21];
  const int* hecl = (const int*)d_in[22];
  const int* hrrw = (const int*)d_in[23];
  const int* hrcl = (const int*)d_in[24];
  const int* trip = (const int*)d_in[25];
  float* out = (float*)d_out;

  char* wp = (char*)d_ws;
  auto walloc = [&](size_t bytes) -> void* {
    void* p = (void*)wp;
    wp += (bytes + 255) & ~((size_t)255);
    return p;
  };
  float* entA = (float*)walloc((size_t)cNE * HID * 4);
  float* entB = (float*)walloc((size_t)cNE * HID * 4);
  float* relA = (float*)walloc((size_t)cR2 * HID * 4);
  float* relB = (float*)walloc((size_t)cR2 * HID * 4);
  float* xsum = (float*)walloc((size_t)cR2 * HID * 4);
  float* xcnt = (float*)walloc((size_t)cR2 * 4);
  float* xin  = (float*)walloc((size_t)cR2 * 2 * HID * 4);
  float* gi   = (float*)walloc((size_t)cR2 * 3 * HID * 4);
  float* gh   = (float*)walloc((size_t)cR2 * 3 * HID * 4);
  float* csum = (float*)walloc((size_t)cNE * HID * 4);   // also hyper-ent y
  float* ccnt = (float*)walloc((size_t)cNE * 4);
  __bf16* aggWbf  = (__bf16*)walloc((size_t)HID * HID * 2);
  __bf16* tgWbf   = (__bf16*)walloc((size_t)HID * HID * 2);
  __bf16* fcWbf   = (__bf16*)walloc((size_t)cKFC * HID * 2);
  __bf16* WihTbf  = (__bf16*)walloc((size_t)2 * HID * 3 * HID * 2);
  __bf16* WhhTbf  = (__bf16*)walloc((size_t)HID * 3 * HID * 2);
  float* feat = (float*)walloc((size_t)cNTT * 3 * HID * 4);
  float* q    = (float*)walloc((size_t)cNTT * HID * 4);
  __bf16* qbf  = (__bf16*)walloc((size_t)cNTT * HID * 2);
  __bf16* entT = (__bf16*)walloc((size_t)HID * cNE * 2);
  float* logz = (float*)walloc((size_t)cNTT * 4);
  float* tsc  = (float*)walloc((size_t)cNTT * 4);

  // one-time weight conversions
  k_f2bf<<<cdiv(HID * HID, 256), 256, 0, stream>>>(aggW, aggWbf, HID * HID);
  k_f2bf<<<cdiv(HID * HID, 256), 256, 0, stream>>>(tgW, tgWbf, HID * HID);
  k_f2bf<<<cdiv((long long)cKFC * HID, 256), 256, 0, stream>>>(fcW, fcWbf, cKFC * HID);
  k_wt2bf<<<cdiv(3 * HID * 2 * HID, 256), 256, 0, stream>>>(Wih, WihTbf, 3 * HID, 2 * HID);
  k_wt2bf<<<cdiv(3 * HID * HID, 256), 256, 0, stream>>>(Whh, WhhTbf, 3 * HID, HID);

  const float* entCur = dyn;  float* entNxt = entA;
  const float* relCur = erel0; float* relNxt = relA;

  for (int t = 0; t < cT; t++) {
    k_zero<<<cdiv(cR2 * HID, 256), 256, 0, stream>>>(xsum, cR2 * HID);
    k_zero<<<cdiv(cR2, 256), 256, 0, stream>>>(xcnt, cR2);
    k_scatter_xmean<<<cdiv((long long)cM * HID, 256), 256, 0, stream>>>(
        entCur, rtoe + (size_t)t * cM, rids + (size_t)t * cM, xsum, xcnt);
    k_build_xin<<<cdiv(cR2 * 2 * HID, 256), 256, 0, stream>>>(relCur, xsum, xcnt, xin);
    k_gemm_bias<<<cdiv(cR2, 16), 256, 16 * 2 * HID * 2, stream>>>(
        xin, cR2, 2 * HID, WihTbf, bih, gi, 3 * HID);
    k_gemm_bias<<<cdiv(cR2, 16), 256, 16 * HID * 2, stream>>>(
        relCur, cR2, HID, WhhTbf, bhh, gh, 3 * HID);
    k_gru_gates<<<cdiv(cR2 * HID, 256), 256, 0, stream>>>(gi, gh, relCur, relNxt);
    k_zero<<<cdiv(cNE * HID, 256), 256, 0, stream>>>(csum, cNE * HID);
    k_zero<<<cdiv(cNE, 256), 256, 0, stream>>>(ccnt, cNE);
    k_msg<<<cE / 16, 256, 0, stream>>>(entCur, relNxt, esrc + (size_t)t * cE,
                                       erel + (size_t)t * cE, edst + (size_t)t * cE,
                                       aggWbf, csum);
    k_count<<<cdiv(cE, 256), 256, 0, stream>>>(edst + (size_t)t * cE, ccnt);
    k_gate<<<cNE / 16, 256, 0, stream>>>(entCur, tgWbf, tgb, csum, ccnt, entNxt);
    relCur = relNxt; relNxt = (relNxt == relA) ? relB : relA;
    entCur = entNxt; entNxt = (entNxt == entA) ? entB : entA;
  }
  // after T=3: entCur==entA, relCur==relA

  // hypergraph on entities (y buffer = csum)
  for (int it = 0; it < 2; it++) {
    k_zero<<<cdiv(cNE * HID, 256), 256, 0, stream>>>(csum, cNE * HID);
    k_hscatter<<<cdiv((long long)cNNZE * HID, 256), 256, 0, stream>>>(
        cNNZE, hev, herw, hecl, entCur, csum);
    k_l2res<<<cNE, 256, 0, stream>>>(csum, entCur, entNxt);
    const float* tmp = entCur; entCur = entNxt; entNxt = (float*)tmp;
  }
  // hypergraph on relations (y buffer = xsum)
  for (int it = 0; it < 2; it++) {
    k_zero<<<cdiv(cR2 * HID, 256), 256, 0, stream>>>(xsum, cR2 * HID);
    k_hscatter<<<cdiv((long long)cNNZR * HID, 256), 256, 0, stream>>>(
        cNNZR, hrv, hrrw, hrcl, relCur, xsum);
    k_l2res<<<cR2, 256, 0, stream>>>(xsum, relCur, relNxt);
    const float* tmp = relCur; relCur = relNxt; relNxt = (float*)tmp;
  }

  // decoder
  k_feat<<<cdiv(cNTT * HID, 256), 256, 0, stream>>>(trip, entCur, relCur, sent, feat);
  k_convfc<<<cNTT / 16, 256, 0, stream>>>(feat, convw, convb, fcWbf, fcb, q, qbf);
  k_entT<<<cdiv(cNE * HID, 256), 256, 0, stream>>>(entCur, entT);
  k_scores<<<cNTT / 16, 256, 0, stream>>>(qbf, entT, logz);
  k_tscore<<<cNTT, 256, 0, stream>>>(q, entCur, trip, tsc);
  k_final<<<1, 256, 0, stream>>>(logz, tsc, out);
}
